// SALSTM_7687991460220
// MI455X (gfx1250) — matrix operand order repaired and checked by reading, not compiled, not run on hardware
//
#include <hip/hip_runtime.h>
#include <stddef.h>

// ---------------------------------------------------------------------------
// Problem constants (from the reference)
// ---------------------------------------------------------------------------
#define SEQ 256
#define BSZ 64
#define ADIM 512
#define INDIM 512
#define HDIM 1024
#define H4 4096

typedef __bf16 bf16_t;
typedef __attribute__((ext_vector_type(16))) __bf16 v16bf;
typedef __attribute__((ext_vector_type(8)))  __bf16 v8bf;
typedef __attribute__((ext_vector_type(8)))  float  v8f;

// ---------------------------------------------------------------------------
// WMMA fragment loaders (wave32, v_wmma_f32_16x16x32_bf16)
//
// A matrix 16x32 bf16: lane<16 -> M=lane, K = 0..15;
// lane>=16 -> M=lane-16, K = 16..31. Two contiguous 16B loads/lane.
// ---------------------------------------------------------------------------
__device__ __forceinline__ v16bf load_a_frag(const bf16_t* __restrict__ base, int lda) {
    const int l = threadIdx.x;                       // 0..31
    const bf16_t* p = base + (size_t)(l & 15) * lda + ((l >> 4) << 4);
    v8bf lo = *(const v8bf*)(p);
    v8bf hi = *(const v8bf*)(p + 8);
    v16bf a;
#pragma unroll
    for (int i = 0; i < 8; ++i) { a[i] = lo[i]; a[8 + i] = hi[i]; }
    return a;
}

// B matrix 32x16 bf16, sourced from a pre-transposed [N,K] weight buffer:
// lane<16 -> N=lane, K=0..15 ; lane>=16 -> N=lane-16, K=16..31.
// One contiguous 32B load per lane.
__device__ __forceinline__ v16bf load_b_frag(const bf16_t* __restrict__ baseT, int ldk) {
    const int l = threadIdx.x;
    const bf16_t* p = baseT + (size_t)(l & 15) * ldk + ((l >> 4) << 4);
    return *(const v16bf*)p;
}

__device__ __forceinline__ v8f wmma_bf16(v16bf a, v16bf b, v8f c) {
    return __builtin_amdgcn_wmma_f32_16x16x32_bf16(false, a, false, b, (short)0, c,
                                                   false, false);
}

__device__ __forceinline__ float sigmoidf_fast(float x) {
    return 1.0f / (1.0f + __expf(-x));
}

// ---------------------------------------------------------------------------
// Conversion / setup kernels
// ---------------------------------------------------------------------------
__global__ void k_to_bf16(const float* __restrict__ in, bf16_t* __restrict__ out, size_t n) {
    size_t i = (size_t)blockIdx.x * blockDim.x + threadIdx.x;
    if (i < n) out[i] = (bf16_t)in[i];
}

__global__ void k_copy_f32(const float* __restrict__ in, float* __restrict__ out, size_t n) {
    size_t i = (size_t)blockIdx.x * blockDim.x + threadIdx.x;
    if (i < n) out[i] = in[i];
}

__global__ void k_add3(const float* __restrict__ a, const float* __restrict__ b,
                       const float* __restrict__ c, float* __restrict__ o, int n) {
    int i = blockIdx.x * blockDim.x + threadIdx.x;
    if (i < n) o[i] = a[i] + b[i] + c[i];
}

// f32 [K,N] row-major  ->  bf16 [N,K] row-major (LDS-tiled transpose)
__global__ void k_transpose_to_bf16(const float* __restrict__ in, bf16_t* __restrict__ out,
                                    int K, int N) {
    __shared__ float tile[32][33];
    const int kb = blockIdx.y * 32;
    const int nb = blockIdx.x * 32;
#pragma unroll
    for (int i = threadIdx.y; i < 32; i += 8) {
        tile[i][threadIdx.x] = in[(size_t)(kb + i) * N + (nb + threadIdx.x)];
    }
    __syncthreads();
#pragma unroll
    for (int i = threadIdx.y; i < 32; i += 8) {
        out[(size_t)(nb + i) * K + (kb + threadIdx.x)] = (bf16_t)tile[threadIdx.x][i];
    }
}

// ---------------------------------------------------------------------------
// comb = a0 @ Wah + (bah + bih + bhh)          M=64, N=4096, K=512
// One wave per 16x16 tile.
// ---------------------------------------------------------------------------
__global__ __launch_bounds__(32)
void k_gemm_comb(const bf16_t* __restrict__ a0bf,  // [64, K]
                 const bf16_t* __restrict__ WT,    // [N, K] (transposed)
                 const float*  __restrict__ bias3, // [N]
                 float* __restrict__ comb,         // [64, N]
                 int N, int K) {
    const int m0 = blockIdx.x * 16;
    const int n0 = blockIdx.y * 16;
    v8f acc = {};
    for (int kk = 0; kk < K; kk += 32) {
        v16bf a = load_a_frag(a0bf + (size_t)m0 * K + kk, K);
        v16bf b = load_b_frag(WT + (size_t)n0 * K + kk, K);
        acc = wmma_bf16(a, b, acc);
    }
    const int l = threadIdx.x;
    const int n = n0 + (l & 15);
    const int mb = m0 + ((l >> 4) << 3);
#pragma unroll
    for (int r = 0; r < 8; ++r) {
        comb[(size_t)(mb + r) * N + n] = acc[r] + bias3[n];
    }
}

// ---------------------------------------------------------------------------
// pre = X @ W + comb[row % B]                  M=SEQ*B, N=4096
// One wave computes a 16(M) x 64(N) strip: 4 accumulators share the A frag.
// ---------------------------------------------------------------------------
__global__ __launch_bounds__(32)
void k_gemm_pre(const bf16_t* __restrict__ Xbf,   // [M, K]
                const bf16_t* __restrict__ WT,    // [N, K]
                const float*  __restrict__ comb,  // [BSZ, N]
                float* __restrict__ out,          // [M, N]
                int N, int K) {
    const int m0 = blockIdx.x * 16;
    const int n0 = blockIdx.y * 64;
    v8f acc[4] = {};
    for (int kk = 0; kk < K; kk += 32) {
        v16bf a = load_a_frag(Xbf + (size_t)m0 * K + kk, K);
#pragma unroll
        for (int j = 0; j < 4; ++j) {
            v16bf b = load_b_frag(WT + (size_t)(n0 + 16 * j) * K + kk, K);
            acc[j] = wmma_bf16(a, b, acc[j]);
        }
    }
    const int l  = threadIdx.x;
    const int ni = (l & 15);
    const int mb = m0 + ((l >> 4) << 3);
#pragma unroll
    for (int j = 0; j < 4; ++j) {
        const int n = n0 + 16 * j + ni;
#pragma unroll
        for (int r = 0; r < 8; ++r) {
            const int m = mb + r;
            out[(size_t)m * N + n] = acc[j][r] + comb[(size_t)(m & (BSZ - 1)) * N + n];
        }
    }
}

// ---------------------------------------------------------------------------
// One recurrent step, fused: gates = pre[t] + h_prev @ Whh ; LSTM update.
// Grid (B/16, H/16), one wave per 16(batch) x 16(hidden) tile.
// Each wave accumulates the 4 gate tiles (i,f,g,o) for its hidden columns,
// so the nonlinearity runs entirely in registers.
// ---------------------------------------------------------------------------
__global__ __launch_bounds__(32)
void k_lstm_step(const bf16_t* __restrict__ hprev,  // [B, H] bf16
                 const bf16_t* __restrict__ WhhT,   // [4H, H] bf16 (transposed)
                 const float*  __restrict__ pre_t,  // [B, 4H]
                 float* __restrict__ cstate,        // [B, H] in-place
                 bf16_t* __restrict__ hnext,        // [B, H] bf16
                 float* __restrict__ hf32,          // nullable [B, H]
                 float* __restrict__ hf32b,         // nullable [B, H]
                 float* __restrict__ cf32) {        // nullable [B, H]
    const int m0 = blockIdx.x * 16;   // batch tile
    const int n0 = blockIdx.y * 16;   // hidden-column tile
    v8f acc[4] = {};
    for (int kk = 0; kk < HDIM; kk += 32) {
        v16bf a = load_a_frag(hprev + (size_t)m0 * HDIM + kk, HDIM);
#pragma unroll
        for (int g = 0; g < 4; ++g) {
            v16bf b = load_b_frag(WhhT + (size_t)(g * HDIM + n0) * HDIM + kk, HDIM);
            acc[g] = wmma_bf16(a, b, acc[g]);
        }
    }
    const int l  = threadIdx.x;
    const int n  = n0 + (l & 15);
    const int mb = m0 + ((l >> 4) << 3);
#pragma unroll
    for (int r = 0; r < 8; ++r) {
        const int m = mb + r;
        const size_t gi = (size_t)m * H4;
        float ig = acc[0][r] + pre_t[gi + 0 * HDIM + n];
        float fg = acc[1][r] + pre_t[gi + 1 * HDIM + n];
        float gg = acc[2][r] + pre_t[gi + 2 * HDIM + n];
        float og = acc[3][r] + pre_t[gi + 3 * HDIM + n];
        const size_t ci = (size_t)m * HDIM + n;
        float cn = sigmoidf_fast(fg) * cstate[ci] + sigmoidf_fast(ig) * tanhf(gg);
        float hn = sigmoidf_fast(og) * tanhf(cn);
        cstate[ci] = cn;
        hnext[ci]  = (bf16_t)hn;
        if (hf32)  hf32[ci]  = hn;
        if (hf32b) hf32b[ci] = hn;
        if (cf32)  cf32[ci]  = cn;
    }
}

// ---------------------------------------------------------------------------
// Host orchestration
// ---------------------------------------------------------------------------
extern "C" void kernel_launch(void* const* d_in, const int* in_sizes, int n_in,
                              void* d_out, int out_size, void* d_ws, size_t ws_size,
                              hipStream_t stream) {
    (void)in_sizes; (void)n_in; (void)out_size; (void)ws_size;

    const float* attention = (const float*)d_in[0];   // [SEQ,B,A]
    const float* x         = (const float*)d_in[1];   // [SEQ,B,IN]
    const float* h0        = (const float*)d_in[2];   // [2,B,H]
    const float* c0        = (const float*)d_in[3];   // [2,B,H]
    const float* Wah0 = (const float*)d_in[4];
    const float* Wih0 = (const float*)d_in[5];
    const float* Whh0 = (const float*)d_in[6];
    const float* bah0 = (const float*)d_in[7];
    const float* bih0 = (const float*)d_in[8];
    const float* bhh0 = (const float*)d_in[9];
    const float* Wah1 = (const float*)d_in[10];
    const float* Wih1 = (const float*)d_in[11];
    const float* Whh1 = (const float*)d_in[12];
    const float* bah1 = (const float*)d_in[13];
    const float* bih1 = (const float*)d_in[14];
    const float* bhh1 = (const float*)d_in[15];

    float* out    = (float*)d_out;
    float* hs1    = out;                               // [SEQ,B,H]
    float* h_last = out + (size_t)SEQ * BSZ * HDIM;    // [2,B,H]
    float* c_last = h_last + 2 * (size_t)BSZ * HDIM;   // [2,B,H]
    const size_t BH = (size_t)BSZ * HDIM;

    // --- carve workspace -------------------------------------------------
    size_t off = 0;
    char* wsb = (char*)d_ws;
    auto carve = [&](size_t bytes) -> void* {
        void* p = wsb + off;
        off += (bytes + 255) & ~(size_t)255;
        return p;
    };
    bf16_t* WahT0 = (bf16_t*)carve((size_t)H4 * ADIM  * 2);
    bf16_t* WihT0 = (bf16_t*)carve((size_t)H4 * INDIM * 2);
    bf16_t* WhhT0 = (bf16_t*)carve((size_t)H4 * HDIM  * 2);
    bf16_t* WahT1 = (bf16_t*)carve((size_t)H4 * ADIM  * 2);
    bf16_t* WihT1 = (bf16_t*)carve((size_t)H4 * HDIM  * 2);
    bf16_t* WhhT1 = (bf16_t*)carve((size_t)H4 * HDIM  * 2);
    bf16_t* xbf   = (bf16_t*)carve((size_t)SEQ * BSZ * INDIM * 2);
    bf16_t* a0bf  = (bf16_t*)carve((size_t)BSZ * ADIM * 2);
    // hs0 chain: slot 0 = h0[layer0], slot t+1 = h at step t   -> [(SEQ+1),B,H]
    bf16_t* hs0   = (bf16_t*)carve((size_t)(SEQ + 1) * BH * 2);
    bf16_t* h1i   = (bf16_t*)carve(BH * 2);            // layer-1 initial h (bf16)
    bf16_t* h1p0  = (bf16_t*)carve(BH * 2);            // layer-1 ping
    bf16_t* h1p1  = (bf16_t*)carve(BH * 2);            // layer-1 pong
    float*  bias30 = (float*)carve((size_t)H4 * 4);
    float*  bias31 = (float*)carve((size_t)H4 * 4);
    float*  comb0  = (float*)carve((size_t)BSZ * H4 * 4);
    float*  comb1  = (float*)carve((size_t)BSZ * H4 * 4);
    float*  cst    = (float*)carve(2 * BH * 4);        // c state, both layers
    float*  pre    = (float*)carve((size_t)SEQ * BSZ * H4 * 4); // reused per layer

    // --- weight transpose+convert (f32 [K,N] -> bf16 [N,K]) --------------
    {
        dim3 tb(32, 8);
        k_transpose_to_bf16<<<dim3(H4 / 32, ADIM  / 32), tb, 0, stream>>>(Wah0, WahT0, ADIM,  H4);
        k_transpose_to_bf16<<<dim3(H4 / 32, INDIM / 32), tb, 0, stream>>>(Wih0, WihT0, INDIM, H4);
        k_transpose_to_bf16<<<dim3(H4 / 32, HDIM  / 32), tb, 0, stream>>>(Whh0, WhhT0, HDIM,  H4);
        k_transpose_to_bf16<<<dim3(H4 / 32, ADIM  / 32), tb, 0, stream>>>(Wah1, WahT1, ADIM,  H4);
        k_transpose_to_bf16<<<dim3(H4 / 32, HDIM  / 32), tb, 0, stream>>>(Wih1, WihT1, HDIM,  H4);
        k_transpose_to_bf16<<<dim3(H4 / 32, HDIM  / 32), tb, 0, stream>>>(Whh1, WhhT1, HDIM,  H4);
    }
    // --- activations / state setup ---------------------------------------
    {
        const size_t nx = (size_t)SEQ * BSZ * INDIM;
        k_to_bf16<<<(int)((nx + 255) / 256), 256, 0, stream>>>(x, xbf, nx);
        const size_t na = (size_t)BSZ * ADIM;            // attention[0]
        k_to_bf16<<<(int)((na + 255) / 256), 256, 0, stream>>>(attention, a0bf, na);
        k_to_bf16<<<(int)((BH + 255) / 256), 256, 0, stream>>>(h0, hs0, BH);        // layer0 h init
        k_to_bf16<<<(int)((BH + 255) / 256), 256, 0, stream>>>(h0 + BH, h1i, BH);   // layer1 h init
        k_copy_f32<<<(int)((2 * BH + 255) / 256), 256, 0, stream>>>(c0, cst, 2 * BH);
        k_add3<<<H4 / 256, 256, 0, stream>>>(bah0, bih0, bhh0, bias30, H4);
        k_add3<<<H4 / 256, 256, 0, stream>>>(bah1, bih1, bhh1, bias31, H4);
    }
    // --- comb = a0 @ Wah + biases  (both layers) --------------------------
    k_gemm_comb<<<dim3(BSZ / 16, H4 / 16), 32, 0, stream>>>(a0bf, WahT0, bias30, comb0, H4, ADIM);
    k_gemm_comb<<<dim3(BSZ / 16, H4 / 16), 32, 0, stream>>>(a0bf, WahT1, bias31, comb1, H4, ADIM);

    // ===================== layer 0 =======================================
    k_gemm_pre<<<dim3(SEQ * BSZ / 16, H4 / 64), 32, 0, stream>>>(
        xbf, WihT0, comb0, pre, H4, INDIM);

    for (int t = 0; t < SEQ; ++t) {
        const bool last = (t == SEQ - 1);
        k_lstm_step<<<dim3(BSZ / 16, HDIM / 16), 32, 0, stream>>>(
            hs0 + (size_t)t * BH, WhhT0, pre + (size_t)t * BSZ * H4,
            cst, hs0 + (size_t)(t + 1) * BH,
            last ? h_last : (float*)nullptr,
            (float*)nullptr,
            last ? c_last : (float*)nullptr);
    }

    // ===================== layer 1 =======================================
    // hs0 slots 1..SEQ form the [SEQ*B, H] activation matrix, already bf16.
    k_gemm_pre<<<dim3(SEQ * BSZ / 16, H4 / 64), 32, 0, stream>>>(
        hs0 + BH, WihT1, comb1, pre, H4, HDIM);

    for (int t = 0; t < SEQ; ++t) {
        const bool last = (t == SEQ - 1);
        const bf16_t* hp = (t == 0) ? h1i : ((t - 1) & 1 ? h1p1 : h1p0);
        bf16_t*       hn = (t & 1) ? h1p1 : h1p0;
        k_lstm_step<<<dim3(BSZ / 16, HDIM / 16), 32, 0, stream>>>(
            hp, WhhT1, pre + (size_t)t * BSZ * H4,
            cst + BH, hn,
            hs1 + (size_t)t * BH,
            last ? (h_last + BH) : (float*)nullptr,
            last ? (c_last + BH) : (float*)nullptr);
    }
}
